// AttentionBlock_76312978915579
// MI455X (gfx1250) — matrix operand order repaired and checked by reading, not compiled, not run on hardware
//
#include <hip/hip_runtime.h>
#include <math.h>

#define B_   16
#define C_   512
#define G_   8
#define NSP  1024      /* H*W */
#define EPSF 1e-5f

typedef __attribute__((ext_vector_type(16))) __bf16 bf16x16;
typedef __attribute__((ext_vector_type(8)))  __bf16 bf16x8;
typedef __attribute__((ext_vector_type(8)))  float  f32x8;

__device__ __forceinline__ __bf16 f2bf(float f) {
    unsigned u = __builtin_bit_cast(unsigned, f);
    u += 0x7FFFu + ((u >> 16) & 1u);          // round-to-nearest-even
    unsigned short h = (unsigned short)(u >> 16);
    return __builtin_bit_cast(__bf16, h);
}

// ---------------------------------------------------------------- convert
__global__ void k_f32_to_bf16(const float* __restrict__ src,
                              __bf16* __restrict__ dst, int n) {
    int i = blockIdx.x * blockDim.x + threadIdx.x;
    if (i < n) dst[i] = f2bf(src[i]);
}

// ---------------------------------------------------------------- groupnorm
// One block per (b, g).  Writes normalized activations TRANSPOSED:
// xn_t[b, n, c] (bf16) so all downstream GEMM B-operands are K-contiguous.
__global__ __launch_bounds__(1024)
void k_groupnorm(const float* __restrict__ x, const float* __restrict__ gamma,
                 const float* __restrict__ beta, __bf16* __restrict__ xn_t) {
    const int b = blockIdx.x / G_;
    const int g = blockIdx.x % G_;
    const int CPG   = C_ / G_;        // 64
    const int total = CPG * NSP;      // 65536
    const float* xb = x + ((size_t)b * C_ + (size_t)g * CPG) * NSP;

    __shared__ float s_sum[1024], s_sq[1024];
    float sum = 0.f, sq = 0.f;
    for (int idx = threadIdx.x; idx < total; idx += 1024) {
        float v = xb[idx];
        sum += v; sq += v * v;
    }
    s_sum[threadIdx.x] = sum; s_sq[threadIdx.x] = sq;
    __syncthreads();
    for (int off = 512; off > 0; off >>= 1) {
        if ((int)threadIdx.x < off) {
            s_sum[threadIdx.x] += s_sum[threadIdx.x + off];
            s_sq[threadIdx.x]  += s_sq[threadIdx.x + off];
        }
        __syncthreads();
    }
    const float inv_n = 1.0f / (float)total;
    const float mean  = s_sum[0] * inv_n;
    const float var   = s_sq[0] * inv_n - mean * mean;
    const float rstd  = rsqrtf(var + EPSF);

    for (int idx = threadIdx.x; idx < total; idx += 1024) {
        int cl = idx >> 10;                // channel within group
        int n  = idx & (NSP - 1);
        int c  = g * CPG + cl;
        float v = (xb[idx] - mean) * rstd * gamma[c] + beta[c];
        xn_t[((size_t)b * NSP + n) * C_ + c] = f2bf(v);
    }
}

// ---------------------------------------------------------------- WMMA GEMM
// D[m,n] = scale * sum_k A[m,k] * Bt[n,k]  (+ bias[m]) (+ res[m,n])
// A: [M,K] row-major bf16 (row stride == K), Bt: [N,K] row-major bf16.
// mode 0: D bf16 [m*N+n]   mode 1: D bf16 transposed [n*M+m]
// mode 2: D f32  [m*N+n] with optional residual add
// 128x128 block tile, 8 waves (4m x 2n), 32x64 per wave = 8 WMMA / K-step.
// Double-buffered LDS, single barrier per K-step, global loads pipelined
// ahead of the WMMA burst.
#define BM 128
#define BN 128
#define BK 32
#define LDSP 40   /* padded LDS row (bf16 elems): 80B -> conflict-free */

__global__ __launch_bounds__(256)
void k_gemm_bf16_wmma(const __bf16* __restrict__ A,  unsigned long long sA,
                      const __bf16* __restrict__ Bt, unsigned long long sB,
                      const float*  __restrict__ bias, float scale,
                      void* __restrict__ Dv, unsigned long long sD,
                      const float* __restrict__ res, unsigned long long sRes,
                      int M, int N, int K, int mode) {
    __shared__ __bf16 As[2][BM * LDSP];   // 2 x 10240 B
    __shared__ __bf16 Bs[2][BN * LDSP];   // 2 x 10240 B

    const int tid  = threadIdx.x;
    const int lane = tid & 31;
    const int wave = tid >> 5;        // 8 waves
    const int wm   = wave & 3;        // 4 m-sub-blocks of 32
    const int wn   = wave >> 2;       // 2 n-sub-blocks of 64
    const int hf   = lane >> 4;       // lane half (wave32)
    const int l15  = lane & 15;
    const int m0   = blockIdx.y * BM;
    const int n0   = blockIdx.x * BN;
    const int bz   = blockIdx.z;
    A  += (size_t)bz * sA;
    Bt += (size_t)bz * sB;

    f32x8 acc[2][4];
#pragma unroll
    for (int i = 0; i < 2; i++)
#pragma unroll
        for (int j = 0; j < 4; j++)
#pragma unroll
            for (int r = 0; r < 8; r++) acc[i][j][r] = 0.f;

    // staging registers (global -> reg -> LDS pipeline)
    bf16x8 ra[2], rb[2];
    const int srow = tid >> 2;        // 0..63 (rows t and t+64)
    const int scc  = (tid & 3) * 8;   // 16B chunk within 32-elem row

    auto gload = [&](int kt) {
#pragma unroll
        for (int rep = 0; rep < 2; rep++) {
            int row = srow + rep * 64;
            ra[rep] = *(const bf16x8*)(A  + (size_t)(m0 + row) * K + kt + scc);
            rb[rep] = *(const bf16x8*)(Bt + (size_t)(n0 + row) * K + kt + scc);
        }
    };
    auto lstore = [&](int buf) {
#pragma unroll
        for (int rep = 0; rep < 2; rep++) {
            int row = srow + rep * 64;
            *(bf16x8*)(&As[buf][row * LDSP + scc]) = ra[rep];
            *(bf16x8*)(&Bs[buf][row * LDSP + scc]) = rb[rep];
        }
    };

    gload(0);
    lstore(0);
    __syncthreads();
    int buf = 0;

    for (int kt = 0; kt < K; kt += BK) {
        const bool hasNext = (kt + BK) < K;
        if (hasNext) gload(kt + BK);   // issue global loads early

        // A frag: lane half 0 holds K 0-7 / 16-23, half 1 holds K 8-15 / 24-31
        bf16x16 afrag[2], bfrag[4];
#pragma unroll
        for (int tm = 0; tm < 2; tm++) {
            const __bf16* ap = &As[buf][(wm * 32 + tm * 16 + l15) * LDSP + hf * 8];
            bf16x8 lo = *(const bf16x8*)ap;
            bf16x8 hi = *(const bf16x8*)(ap + 16);
            afrag[tm] = __builtin_shufflevector(lo, hi,
                0,1,2,3,4,5,6,7,8,9,10,11,12,13,14,15);
        }
        // B frag: lane half 0 holds K 0-7 / 16-23, half 1 holds K 8-15 / 24-31 of row n
#pragma unroll
        for (int tn = 0; tn < 4; tn++) {
            const __bf16* bpp = &Bs[buf][(wn * 64 + tn * 16 + l15) * LDSP + hf * 8];
            bf16x8 lo = *(const bf16x8*)bpp;
            bf16x8 hi = *(const bf16x8*)(bpp + 16);
            bfrag[tn] = __builtin_shufflevector(lo, hi,
                0,1,2,3,4,5,6,7,8,9,10,11,12,13,14,15);
        }
#pragma unroll
        for (int tm = 0; tm < 2; tm++)
#pragma unroll
            for (int tn = 0; tn < 4; tn++)
                acc[tm][tn] = __builtin_amdgcn_wmma_f32_16x16x32_bf16(
                    false, afrag[tm], false, bfrag[tn],
                    (short)0, acc[tm][tn], false, false);

        if (hasNext) {
            lstore(buf ^ 1);           // fill the other buffer
            __syncthreads();           // one barrier per K-step
            buf ^= 1;
        }
    }

    // epilogue: D lane map m = 8*half + r, n = lane&15
#pragma unroll
    for (int tm = 0; tm < 2; tm++) {
#pragma unroll
        for (int tn = 0; tn < 4; tn++) {
            int nn = n0 + wn * 64 + tn * 16 + l15;
#pragma unroll
            for (int r = 0; r < 8; r++) {
                int mm = m0 + wm * 32 + tm * 16 + hf * 8 + r;
                float v = acc[tm][tn][r] * scale;
                if (bias) v += bias[mm];
                if (mode == 0) {
                    __bf16* D = (__bf16*)Dv + (size_t)bz * sD;
                    D[(size_t)mm * N + nn] = f2bf(v);
                } else if (mode == 1) {
                    __bf16* D = (__bf16*)Dv + (size_t)bz * sD;
                    D[(size_t)nn * M + mm] = f2bf(v);
                } else {
                    float* D = (float*)Dv + (size_t)bz * sD;
                    if (res) v += res[(size_t)bz * sRes + (size_t)mm * N + nn];
                    D[(size_t)mm * N + nn] = v;
                }
            }
        }
    }
}

// ---------------------------------------------------------------- softmax
// One block per attention row (b,i): 1024 f32 scores -> bf16 probabilities.
__global__ __launch_bounds__(256)
void k_softmax_rows(const float* __restrict__ S, __bf16* __restrict__ P) {
    const size_t row = blockIdx.x;
    const float* s = S + row * NSP;
    __shared__ float red[256];
    const int t = threadIdx.x;

    float v[4];
    float mx = -INFINITY;
#pragma unroll
    for (int i = 0; i < 4; i++) { v[i] = s[t + i * 256]; mx = fmaxf(mx, v[i]); }
    red[t] = mx; __syncthreads();
    for (int off = 128; off > 0; off >>= 1) {
        if (t < off) red[t] = fmaxf(red[t], red[t + off]);
        __syncthreads();
    }
    const float m = red[0];
    __syncthreads();

    float sum = 0.f;
#pragma unroll
    for (int i = 0; i < 4; i++) { v[i] = __expf(v[i] - m); sum += v[i]; }
    red[t] = sum; __syncthreads();
    for (int off = 128; off > 0; off >>= 1) {
        if (t < off) red[t] += red[t + off];
        __syncthreads();
    }
    const float inv = 1.0f / red[0];
#pragma unroll
    for (int i = 0; i < 4; i++)
        P[row * NSP + t + i * 256] = f2bf(v[i] * inv);
}

// ---------------------------------------------------------------- launch
extern "C" void kernel_launch(void* const* d_in, const int* in_sizes, int n_in,
                              void* d_out, int out_size, void* d_ws, size_t ws_size,
                              hipStream_t stream) {
    const float* x     = (const float*)d_in[0];
    const float* gamma = (const float*)d_in[1];
    const float* beta  = (const float*)d_in[2];
    const float* wq    = (const float*)d_in[3];
    const float* bq    = (const float*)d_in[4];
    const float* wk    = (const float*)d_in[5];
    const float* bk    = (const float*)d_in[6];
    const float* wv    = (const float*)d_in[7];
    const float* bv    = (const float*)d_in[8];
    const float* wp    = (const float*)d_in[9];
    const float* bp    = (const float*)d_in[10];
    float* out = (float*)d_out;

    // workspace carve
    const size_t szW   = (size_t)C_ * C_ * 2;          // bf16 weight
    const size_t szBNC = (size_t)B_ * NSP * C_ * 2;    // bf16 [B,N,C] / [B,C,N]
    const size_t szS   = (size_t)B_ * NSP * NSP * 4;   // f32 scores
    char* w = (char*)d_ws; size_t off = 0;
    __bf16* wqb  = (__bf16*)(w + off); off += szW;
    __bf16* wkb  = (__bf16*)(w + off); off += szW;
    __bf16* wvb  = (__bf16*)(w + off); off += szW;
    __bf16* wpb  = (__bf16*)(w + off); off += szW;
    __bf16* xn_t = (__bf16*)(w + off); off += szBNC;   // [B,N,C]
    __bf16* q_t  = (__bf16*)(w + off); off += szBNC;   // [B,N,C]
    __bf16* k_t  = (__bf16*)(w + off); off += szBNC;   // [B,N,C]
    __bf16* vv   = (__bf16*)(w + off); off += szBNC;   // [B,C,N]
    float*  sS   = (float*)(w + off);  off += szS;     // [B,N,N]
    __bf16* attn = (__bf16*)(w + off); off += (size_t)B_ * NSP * NSP * 2; // [B,N,N]
    __bf16* o_t  = (__bf16*)(w + off); off += szBNC;   // [B,N,C]

    const int nW = C_ * C_;
    k_f32_to_bf16<<<(nW + 255) / 256, 256, 0, stream>>>(wq, wqb, nW);
    k_f32_to_bf16<<<(nW + 255) / 256, 256, 0, stream>>>(wk, wkb, nW);
    k_f32_to_bf16<<<(nW + 255) / 256, 256, 0, stream>>>(wv, wvb, nW);
    k_f32_to_bf16<<<(nW + 255) / 256, 256, 0, stream>>>(wp, wpb, nW);

    k_groupnorm<<<B_ * G_, 1024, 0, stream>>>(x, gamma, beta, xn_t);

    const unsigned long long sBNC = (unsigned long long)NSP * C_;
    const unsigned long long sBNN = (unsigned long long)NSP * NSP;

    // Q = Wq @ Xn  -> q_t[b,n,c]  (transposed store)
    k_gemm_bf16_wmma<<<dim3(NSP / BN, C_ / BM, B_), 256, 0, stream>>>(
        wqb, 0ULL, xn_t, sBNC, bq, 1.0f, q_t, sBNC, nullptr, 0ULL,
        C_, NSP, C_, 1);
    // K = Wk @ Xn  -> k_t[b,n,c]
    k_gemm_bf16_wmma<<<dim3(NSP / BN, C_ / BM, B_), 256, 0, stream>>>(
        wkb, 0ULL, xn_t, sBNC, bk, 1.0f, k_t, sBNC, nullptr, 0ULL,
        C_, NSP, C_, 1);
    // V = Wv @ Xn  -> v[b,c,n]   (natural store)
    k_gemm_bf16_wmma<<<dim3(NSP / BN, C_ / BM, B_), 256, 0, stream>>>(
        wvb, 0ULL, xn_t, sBNC, bv, 1.0f, vv, sBNC, nullptr, 0ULL,
        C_, NSP, C_, 0);
    // S = scale * Q^T K : A = q_t[i,c], Bt = k_t[j,c]  -> f32 [b,i,j]
    const float scl = 1.0f / sqrtf((float)C_);
    k_gemm_bf16_wmma<<<dim3(NSP / BN, NSP / BM, B_), 256, 0, stream>>>(
        q_t, sBNC, k_t, sBNC, nullptr, scl, sS, sBNN, nullptr, 0ULL,
        NSP, NSP, C_, 2);

    k_softmax_rows<<<B_ * NSP, 256, 0, stream>>>(sS, attn);

    // O = V @ attn^T : A = v[c,j], Bt = attn[i,j] -> o_t[b,i,c]
    k_gemm_bf16_wmma<<<dim3(NSP / BN, C_ / BM, B_), 256, 0, stream>>>(
        vv, sBNC, attn, sBNN, nullptr, 1.0f, o_t, sBNC, nullptr, 0ULL,
        C_, NSP, NSP, 1);
    // out = x + Wp @ O + bp : A = wp, Bt = o_t -> f32 [b,c,n] + residual
    k_gemm_bf16_wmma<<<dim3(NSP / BN, C_ / BM, B_), 256, 0, stream>>>(
        wpb, 0ULL, o_t, sBNC, bp, 1.0f, out, sBNC, x, sBNC,
        C_, NSP, C_, 2);
}
